// LSTMPredictor_37220186587538
// MI455X (gfx1250) — matrix operand order repaired and checked by reading, not compiled, not run on hardware
//
#include <hip/hip_runtime.h>
#include <hip/hip_bf16.h>
#include <math.h>

// ---------------------------------------------------------------------------
// LSTMPredictor for MI455X (gfx1250, wave32, WMMA 16x16x32 f16 -> f32 acc)
//
// Pipeline:
//   1) convert weights f32 -> f16 (WMMA operand layout is plain row-major)
//   2) GEMM  xp  = seq  @ w_ih0^T + (b_ih0+b_hh0)        [WMMA, 16x64/wave]
//   3) scan layer0: per-step h@w_hh0^T via WMMA, W_hh resident in VGPRs,
//      xp loads hoisted + L2 prefetch, single b128 gate store per wave,
//      branchless activations (exp + rcp only)
//   4) GEMM  xp  = h1 @ w_ih1^T + (b_ih1+b_hh1)          (xp reused)
//   5) scan layer1
//   6) GEMM  out = h2 @ w_lin^T + b_lin
// ---------------------------------------------------------------------------

typedef __attribute__((ext_vector_type(16))) _Float16 v16h;
typedef __attribute__((ext_vector_type(8)))  _Float16 v8h;
typedef __attribute__((ext_vector_type(8)))  float    v8f;
typedef __attribute__((ext_vector_type(4)))  float    v4f;

#define SEQ_LEN    32768
#define INPUT_DIM  512
#define HIDDEN     128
#define GATES      512   /* 4*HIDDEN */
#define OUT_DIM    256
#define XP_PREFETCH 24   /* steps ahead to warm L2 for the xp stream */

// Branchless activations for the serial critical path.
__device__ __forceinline__ float sigmoid_fast(float x) {
  return __builtin_amdgcn_rcpf(1.0f + __expf(-x));
}
__device__ __forceinline__ float tanh_fast(float x) {
  // 1 - 2/(e^{2x}+1): exp overflow -> +1, underflow -> -1 (no branches)
  return 1.0f - 2.0f * __builtin_amdgcn_rcpf(__expf(2.0f * x) + 1.0f);
}

// ---------------------------------------------------------------------------
// f32 -> f16 conversion
// ---------------------------------------------------------------------------
__global__ void cvt_f32_to_f16(const float* __restrict__ in,
                               _Float16* __restrict__ out, int n) {
  int i = blockIdx.x * blockDim.x + threadIdx.x;
  if (i < n) out[i] = (_Float16)in[i];
}

// ---------------------------------------------------------------------------
// WMMA GEMM:  out[M,N] (f32) = A[M,K] @ W[N,K]^T + b0[N] (+ b1[N])
// One wave per 16x64 output strip (4 N-tiles share each A tile load).
// A is f32 (converted on the fly) or f16. M%16==0, N%64==0, K%32==0.
// ---------------------------------------------------------------------------
template <bool A_IS_F32>
__global__ __launch_bounds__(256)
void gemm_wmma(const void* __restrict__ Av, const _Float16* __restrict__ W,
               const float* __restrict__ b0, const float* __restrict__ b1,
               float* __restrict__ out, int M, int N, int K) {
  const int gtid   = blockIdx.x * blockDim.x + threadIdx.x;
  const int wave   = gtid >> 5;
  const int lane   = threadIdx.x & 31;
  const int l16    = lane & 15;
  const int halfid = lane >> 4;

  const int nclust = N >> 6;                   // 64-wide column clusters
  const int mt = wave / nclust;
  const int nc = wave - mt * nclust;
  if (mt * 16 >= M) return;                    // wave-uniform exit

  const int arow = mt * 16 + l16;              // A row this lane feeds
  const int col0 = nc * 64 + l16;              // first column this lane holds

  v8f acc[4] = {{}, {}, {}, {}};
  for (int kc = 0; kc < K; kc += 32) {
    // ---- A tile 16x32 (f16 lane layout: K = halfid*16 + {0..15}) ----
    v16h a;
    if (A_IS_F32) {
      const float* p = (const float*)Av + (size_t)arow * K + kc + halfid * 16;
      v4f x0 = *(const v4f*)(p);
      v4f x1 = *(const v4f*)(p + 4);
      v4f x2 = *(const v4f*)(p + 8);
      v4f x3 = *(const v4f*)(p + 12);
      v8f lo = __builtin_shufflevector(x0, x1, 0, 1, 2, 3, 4, 5, 6, 7);
      v8f hi = __builtin_shufflevector(x2, x3, 0, 1, 2, 3, 4, 5, 6, 7);
      v8h l8 = __builtin_convertvector(lo, v8h);
      v8h h8 = __builtin_convertvector(hi, v8h);
      a = __builtin_shufflevector(l8, h8, 0, 1, 2, 3, 4, 5, 6, 7,
                                          8, 9, 10, 11, 12, 13, 14, 15);
    } else {
      const _Float16* p = (const _Float16*)Av + (size_t)arow * K + kc + halfid * 16;
      v8h l8 = *(const v8h*)(p);
      v8h h8 = *(const v8h*)(p + 8);
      a = __builtin_shufflevector(l8, h8, 0, 1, 2, 3, 4, 5, 6, 7,
                                          8, 9, 10, 11, 12, 13, 14, 15);
    }
    // ---- 4 B tiles 32x16: lane holds column, K = halfid*16 + {0..15} ----
#pragma unroll
    for (int j = 0; j < 4; ++j) {
      v16h b = *(const v16h*)(W + (size_t)(col0 + j * 16) * K + kc + halfid * 16);
      acc[j] = __builtin_amdgcn_wmma_f32_16x16x32_f16(
          false, a, false, b, (short)0, acc[j], false, false);
    }
  }

#pragma unroll
  for (int j = 0; j < 4; ++j) {
    const int col = col0 + j * 16;
    float bias = b0[col];
    if (b1) bias += b1[col];
#pragma unroll
    for (int r = 0; r < 8; ++r) {
      int orow = mt * 16 + halfid * 8 + r;     // C/D layout: VGPR r -> row
      out[(size_t)orow * N + col] = acc[j][r] + bias;
    }
  }
}

// ---------------------------------------------------------------------------
// LSTM recurrence. Single 256-thread block (8 waves).
//   gates[t] = xp[t] + h @ W_hh^T      (W_hh resident in VGPRs as B-tiles)
// Wave w owns gate columns [w*64, w*64+64): 4 N-tiles x 4 K-chunks of WMMA.
// Gate scratch layout: gates[w*64 + lane*4 + nt] holds column w*64+nt*16+lane
// so each wave emits ONE ds_store_b128 per step.
// nt-outer / kc-inner WMMA ordering keeps all 4 A tiles live so the 8 A
// ds_load_b128 issue upfront with staggered dscnt waits (loads overlap WMMA).
// ---------------------------------------------------------------------------
__device__ __forceinline__ int gate_idx(int c) {
  // column c = w*64 + nt*16 + l  ->  index w*64 + l*4 + nt
  return (c & ~63) | ((c & 15) << 2) | ((c >> 4) & 3);
}

__global__ __launch_bounds__(256)
void lstm_scan(const float* __restrict__ xp,      // [T, 512] pre-activations
               const _Float16* __restrict__ whh,  // [512, 128] row-major
               _Float16* __restrict__ hout,       // [T, 128] f16 hidden out
               int T) {
  __shared__ __align__(32) _Float16 h_sh[HIDDEN];
  __shared__ __align__(16) float gates[GATES];

  const int tid    = threadIdx.x;
  const int lane   = tid & 31;
  const int w      = tid >> 5;                   // wave id 0..7
  const int l16    = lane & 15;
  const int halfid = lane >> 4;

  // ---- Preload this wave's 64 columns of W_hh as 16 B-tiles (128 VGPRs) ---
  v16h btile[16];
#pragma unroll
  for (int nt = 0; nt < 4; ++nt) {
#pragma unroll
    for (int kc = 0; kc < 4; ++kc) {
      const int col = w * 64 + nt * 16 + l16;
      btile[nt * 4 + kc] =
          *(const v16h*)(whh + (size_t)col * HIDDEN + kc * 32 + halfid * 16);
    }
  }

  // precomputed indices for the update phase (threads 0..127)
  const int gi0 = gate_idx(tid);
  const int gi1 = gate_idx(HIDDEN + tid);
  const int gi2 = gate_idx(2 * HIDDEN + tid);
  const int gi3 = gate_idx(3 * HIDDEN + tid);

  if (tid < HIDDEN) h_sh[tid] = (_Float16)0.0f;
  float c = 0.0f;
  __syncthreads();

  for (int t = 0; t < T; ++t) {
    // ---- Issue this step's xp loads early; prefetch the stream into L2 ----
    float x0 = 0.f, x1 = 0.f, x2 = 0.f, x3 = 0.f;
    if (tid < HIDDEN) {
      const float* x = xp + (size_t)t * GATES;
      x0 = x[tid];
      x1 = x[HIDDEN + tid];
      x2 = x[2 * HIDDEN + tid];
      x3 = x[3 * HIDDEN + tid];
      int tp = t + XP_PREFETCH;
      if (tp >= T) tp = T - 1;
      __builtin_prefetch(xp + (size_t)tp * GATES + tid, 0, 1);
    }

    // ---- Broadcast h into A layout: every lane carries row-0 data ----
    v16h a[4];
#pragma unroll
    for (int kc = 0; kc < 4; ++kc) {
      const _Float16* p = h_sh + kc * 32 + halfid * 16;
      v8h l8 = *(const v8h*)(p);
      v8h h8 = *(const v8h*)(p + 8);
      a[kc] = __builtin_shufflevector(l8, h8, 0, 1, 2, 3, 4, 5, 6, 7,
                                              8, 9, 10, 11, 12, 13, 14, 15);
    }

    // ---- h @ W_hh^T : nt-outer so all 4 A tiles stay live ----
    v8f acc[4] = {{}, {}, {}, {}};
#pragma unroll
    for (int nt = 0; nt < 4; ++nt) {
#pragma unroll
      for (int kc = 0; kc < 4; ++kc) {
        acc[nt] = __builtin_amdgcn_wmma_f32_16x16x32_f16(
            false, a[kc], false, btile[nt * 4 + kc], (short)0, acc[nt],
            false, false);
      }
    }

    // ---- One 16-byte gate store per wave (row 0 of each tile) ----
    if (lane < 16) {
      v4f g4 = {acc[0][0], acc[1][0], acc[2][0], acc[3][0]};
      *(v4f*)(gates + w * 64 + lane * 4) = g4;
    }
    __syncthreads();

    // ---- Gate nonlinearity + state update (threads 0..127) ----
    if (tid < HIDDEN) {
      float gi = gates[gi0] + x0;
      float gf = gates[gi1] + x1;
      float gg = gates[gi2] + x2;
      float go = gates[gi3] + x3;
      float i_ = sigmoid_fast(gi);
      float f_ = sigmoid_fast(gf);
      float g_ = tanh_fast(gg);
      float o_ = sigmoid_fast(go);
      c = f_ * c + i_ * g_;
      float h = o_ * tanh_fast(c);
      h_sh[tid] = (_Float16)h;
      hout[(size_t)t * HIDDEN + tid] = (_Float16)h;
    }
    __syncthreads();
  }
}

// ---------------------------------------------------------------------------
// Host-side orchestration
// ---------------------------------------------------------------------------
extern "C" void kernel_launch(void* const* d_in, const int* in_sizes, int n_in,
                              void* d_out, int out_size, void* d_ws, size_t ws_size,
                              hipStream_t stream) {
  const float* seq   = (const float*)d_in[0];
  const float* wih0  = (const float*)d_in[1];
  const float* whh0  = (const float*)d_in[2];
  const float* bih0  = (const float*)d_in[3];
  const float* bhh0  = (const float*)d_in[4];
  const float* wih1  = (const float*)d_in[5];
  const float* whh1  = (const float*)d_in[6];
  const float* bih1  = (const float*)d_in[7];
  const float* bhh1  = (const float*)d_in[8];
  const float* wlin  = (const float*)d_in[9];
  const float* blin  = (const float*)d_in[10];

  // -------- workspace layout --------
  char* p = (char*)d_ws;
  auto take = [&](size_t bytes) {
    char* r = p;
    p += (bytes + 255) & ~(size_t)255;
    return r;
  };
  _Float16* wih0_h = (_Float16*)take((size_t)GATES * INPUT_DIM * 2);  // 512x512
  _Float16* whh0_h = (_Float16*)take((size_t)GATES * HIDDEN * 2);     // 512x128
  _Float16* wih1_h = (_Float16*)take((size_t)GATES * HIDDEN * 2);
  _Float16* whh1_h = (_Float16*)take((size_t)GATES * HIDDEN * 2);
  _Float16* wlin_h = (_Float16*)take((size_t)OUT_DIM * HIDDEN * 2);   // 256x128
  _Float16* h1_h   = (_Float16*)take((size_t)SEQ_LEN * HIDDEN * 2);
  _Float16* h2_h   = (_Float16*)take((size_t)SEQ_LEN * HIDDEN * 2);
  float*    xp     = (float*)take((size_t)SEQ_LEN * GATES * 4);       // reused

  // -------- 1) convert weights to f16 --------
  auto cvt = [&](const float* src, _Float16* dst, int n) {
    cvt_f32_to_f16<<<(n + 255) / 256, 256, 0, stream>>>(src, dst, n);
  };
  cvt(wih0, wih0_h, GATES * INPUT_DIM);
  cvt(whh0, whh0_h, GATES * HIDDEN);
  cvt(wih1, wih1_h, GATES * HIDDEN);
  cvt(whh1, whh1_h, GATES * HIDDEN);
  cvt(wlin, wlin_h, OUT_DIM * HIDDEN);

  // -------- 2) xp = seq @ w_ih0^T + b_ih0 + b_hh0 --------
  {
    int waves = (SEQ_LEN / 16) * (GATES / 64);
    gemm_wmma<true><<<waves * 32 / 256, 256, 0, stream>>>(
        (const void*)seq, wih0_h, bih0, bhh0, xp, SEQ_LEN, GATES, INPUT_DIM);
  }
  // -------- 3) layer-0 recurrence --------
  lstm_scan<<<1, 256, 0, stream>>>(xp, whh0_h, h1_h, SEQ_LEN);

  // -------- 4) xp = h1 @ w_ih1^T + b_ih1 + b_hh1 (xp reused) --------
  {
    int waves = (SEQ_LEN / 16) * (GATES / 64);
    gemm_wmma<false><<<waves * 32 / 256, 256, 0, stream>>>(
        (const void*)h1_h, wih1_h, bih1, bhh1, xp, SEQ_LEN, GATES, HIDDEN);
  }
  // -------- 5) layer-1 recurrence --------
  lstm_scan<<<1, 256, 0, stream>>>(xp, whh1_h, h2_h, SEQ_LEN);

  // -------- 6) out = h2 @ w_lin^T + b_lin --------
  {
    int waves = (SEQ_LEN / 16) * (OUT_DIM / 64);
    gemm_wmma<false><<<waves * 32 / 256, 256, 0, stream>>>(
        (const void*)h2_h, wlin_h, blin, (const float*)nullptr,
        (float*)d_out, SEQ_LEN, OUT_DIM, HIDDEN);
  }
}
